// MultiHeadAttention_37383395345165
// MI455X (gfx1250) — matrix operand order repaired and checked by reading, not compiled, not run on hardware
//
#include <hip/hip_runtime.h>
#include <hip/hip_bf16.h>

#define D_MODEL 1024
#define NUM_HEADS 16
#define HEAD_DIM 64
#define BATCH 4
#define SEQ 2048
#define NTOK (BATCH * SEQ) /* 8192 */

typedef __attribute__((ext_vector_type(16))) _Float16 v16h;
typedef __attribute__((ext_vector_type(8)))  _Float16 v8h;
typedef __attribute__((ext_vector_type(8)))  float    v8f;

__device__ __forceinline__ v16h cat16(v8h lo, v8h hi) {
  return __builtin_shufflevector(lo, hi, 0,1,2,3,4,5,6,7,8,9,10,11,12,13,14,15);
}

// A fragment (16x32 f16) from row-major MxK source, leading dim ld.
// lane L: row = L&15; elems 0-15: k = (L>>4)*16 + 0..15 (contiguous in memory)
__device__ __forceinline__ v16h load_a_frag(const _Float16* base, int ld, int lane) {
  const int row = lane & 15, g = (lane >> 4) & 1;
  const _Float16* p = base + (size_t)row * ld + g * 16;
  v8h lo = *(const v8h*)p;
  v8h hi = *(const v8h*)(p + 8);
  return cat16(lo, hi);
}

// B fragment (32x16) from a row-major N x K source (i.e. B[k][n] = src[n][k]).
// lane L: col n = L&15; elems 0-15: k = (L>>4)*16 + 0..15 (contiguous in memory)
__device__ __forceinline__ v16h load_bt_frag(const _Float16* base, int ld, int lane) {
  const int col = lane & 15, g = (lane >> 4) & 1;
  const _Float16* p = base + (size_t)col * ld + g * 16;
  v8h lo = *(const v8h*)p;
  v8h hi = *(const v8h*)(p + 8);
  return cat16(lo, hi);
}

#define WMMA_F16(a, b, c) \
  __builtin_amdgcn_wmma_f32_16x16x32_f16(false, (a), false, (b), (short)0, (c), false, false)

// ---------------------------------------------------------------------------
__global__ void cvt_f32_f16(const float* __restrict__ src,
                            _Float16* __restrict__ dst, int n) {
  int i = blockIdx.x * blockDim.x + threadIdx.x;
  const int stride = gridDim.x * blockDim.x;
  for (; i < n; i += stride) dst[i] = (_Float16)src[i];
}

// ---------------------------------------------------------------------------
// Fused QKV projection + RoPE. One wave computes a 16(M) x 64(N) tile; the
// 64-col tile is exactly one head. z=0:Q(+rope), 1:K(+rope), 2:V (transposed).
__global__ void qkv_rope_kernel(const _Float16* __restrict__ xh,
                                const _Float16* __restrict__ wq,
                                const _Float16* __restrict__ wk,
                                const _Float16* __restrict__ wv,
                                const int* __restrict__ tp,
                                _Float16* __restrict__ Qh,
                                _Float16* __restrict__ Kh,
                                _Float16* __restrict__ Vt) {
  const int lane = threadIdx.x;
  const int which = blockIdx.z;
  const _Float16* W = (which == 0) ? wq : ((which == 1) ? wk : wv);
  const int m0 = blockIdx.y * 16;
  const int n0 = blockIdx.x * 64;

  v8f acc[4] = {};
  for (int k0 = 0; k0 < D_MODEL; k0 += 32) {
    const v16h a = load_a_frag(xh + (size_t)m0 * D_MODEL + k0, D_MODEL, lane);
#pragma unroll
    for (int j = 0; j < 4; ++j) {
      const v16h b = load_bt_frag(W + (size_t)(n0 + j * 16) * D_MODEL + k0, D_MODEL, lane);
      acc[j] = WMMA_F16(a, b, acc[j]);
    }
  }

  const int nb = m0 >> 11;          // batch
  const int s0 = m0 & (SEQ - 1);    // seq base
  const int h  = n0 >> 6;           // head
  const int hg = lane >> 4;         // lane half-group
  const int cb = lane & 15;         // N within 16-subtile

  if (which < 2) {
    _Float16* Dst = (which == 0) ? Qh : Kh;
#pragma unroll
    for (int j = 0; j < 4; ++j) {
#pragma unroll
      for (int r = 0; r < 8; ++r) {
        const int srow = s0 + r + 8 * hg;
        const int d = j * 16 + cb;
        const float pos = (float)tp[srow];
        // inv_freq = theta^(-2*(d/2)/64) = exp(-(d/2) * ln(10000)/32)
        const float ang = pos * __expf((float)(d >> 1) * -0.2878231366242557f);
        float c, sn;
        __sincosf(ang, &sn, &c);
        const float x = acc[j][r];
        const float p = __shfl_xor(x, 1);          // pair partner (adjacent lane)
        const float y = x * c + ((d & 1) ? (p * sn) : -(p * sn));
        Dst[(((size_t)(nb * NUM_HEADS + h)) * SEQ + srow) * HEAD_DIM + d] = (_Float16)y;
      }
    }
  } else {
    // V stored transposed: Vt[(nh*64 + d) * SEQ + s]  -> contiguous B-frags later
#pragma unroll
    for (int j = 0; j < 4; ++j) {
      v8h pk;
#pragma unroll
      for (int r = 0; r < 8; ++r) pk[r] = (_Float16)acc[j][r];
      const int d = j * 16 + cb;
      *(v8h*)(Vt + (((size_t)(nb * NUM_HEADS + h)) * HEAD_DIM + d) * SEQ + s0 + 8 * hg) = pk;
    }
  }
}

// ---------------------------------------------------------------------------
// Flash attention: one wave per (n,h, 16-query tile). Causal, online softmax.
__global__ void attn_kernel(const _Float16* __restrict__ Qh,
                            const _Float16* __restrict__ Kh,
                            const _Float16* __restrict__ Vt,
                            _Float16* __restrict__ attnh) {
  __shared__ _Float16 pT[16 * 32];
  const int lane = threadIdx.x;
  const int nh = blockIdx.y;
  const int nb = nh >> 4, h = nh & 15;
  const int q0 = blockIdx.x * 16;
  const int hg = lane >> 4, cb = lane & 15;

  const _Float16* Qp = Qh + (size_t)nh * SEQ * HEAD_DIM;
  const _Float16* Kp = Kh + (size_t)nh * SEQ * HEAD_DIM;
  const _Float16* Vp = Vt + (size_t)nh * HEAD_DIM * SEQ;

  const v16h qa0 = load_a_frag(Qp + (size_t)q0 * HEAD_DIM + 0,  HEAD_DIM, lane);
  const v16h qa1 = load_a_frag(Qp + (size_t)q0 * HEAD_DIM + 32, HEAD_DIM, lane);

  v8f o[4] = {};
  float mrow[8], lrow[8];
#pragma unroll
  for (int r = 0; r < 8; ++r) { mrow[r] = -__builtin_inff(); lrow[r] = 0.f; }

  const int nkb = (q0 >> 5) + 1;  // key blocks of 32 needed for causal rows
  for (int kb = 0; kb < nkb; ++kb) {
    const int kbase = kb * 32;
    v8f s0v = {}, s1v = {};
    {
      const v16h b00 = load_bt_frag(Kp + (size_t)(kbase     ) * HEAD_DIM + 0,  HEAD_DIM, lane);
      const v16h b01 = load_bt_frag(Kp + (size_t)(kbase     ) * HEAD_DIM + 32, HEAD_DIM, lane);
      const v16h b10 = load_bt_frag(Kp + (size_t)(kbase + 16) * HEAD_DIM + 0,  HEAD_DIM, lane);
      const v16h b11 = load_bt_frag(Kp + (size_t)(kbase + 16) * HEAD_DIM + 32, HEAD_DIM, lane);
      s0v = WMMA_F16(qa0, b00, s0v); s0v = WMMA_F16(qa1, b01, s0v);
      s1v = WMMA_F16(qa0, b10, s1v); s1v = WMMA_F16(qa1, b11, s1v);
    }
    const float sc = 0.125f;  // 1/sqrt(64)
#pragma unroll
    for (int r = 0; r < 8; ++r) {
      const int qrow = q0 + r + 8 * hg;
      float a0 = s0v[r] * sc, a1 = s1v[r] * sc;
      if (kbase + cb      > qrow) a0 = -__builtin_inff();
      if (kbase + 16 + cb > qrow) a1 = -__builtin_inff();
      float mx = fmaxf(a0, a1);
#pragma unroll
      for (int off = 1; off < 16; off <<= 1) mx = fmaxf(mx, __shfl_xor(mx, off));
      const float mnew = fmaxf(mrow[r], mx);
      float scale, p0, p1;
      if (mnew == -__builtin_inff()) { scale = 1.f; p0 = 0.f; p1 = 0.f; }
      else {
        scale = __expf(mrow[r] - mnew);
        p0 = __expf(a0 - mnew);
        p1 = __expf(a1 - mnew);
      }
      mrow[r] = mnew;
      float ps = p0 + p1;
#pragma unroll
      for (int off = 1; off < 16; off <<= 1) ps += __shfl_xor(ps, off);
      lrow[r] = lrow[r] * scale + ps;
#pragma unroll
      for (int j = 0; j < 4; ++j) o[j][r] *= scale;
      pT[(r + 8 * hg) * 32 + cb     ] = (_Float16)p0;
      pT[(r + 8 * hg) * 32 + 16 + cb] = (_Float16)p1;
    }
    __syncthreads();
    const v16h pa = load_a_frag(pT, 32, lane);   // C-layout -> A-frag via LDS
#pragma unroll
    for (int j = 0; j < 4; ++j) {
      const v16h vb = load_bt_frag(Vp + (size_t)(j * 16) * SEQ + kbase, SEQ, lane);
      o[j] = WMMA_F16(pa, vb, o[j]);
    }
    __syncthreads();
  }

  // normalize, store row-major (n, s, h*64 + d) for the output projection
#pragma unroll
  for (int r = 0; r < 8; ++r) {
    const float inv = 1.f / lrow[r];
    const int srow = q0 + r + 8 * hg;
#pragma unroll
    for (int j = 0; j < 4; ++j) {
      attnh[((size_t)(nb * SEQ + srow)) * D_MODEL + h * HEAD_DIM + j * 16 + cb] =
          (_Float16)(o[j][r] * inv);
    }
  }
}

// ---------------------------------------------------------------------------
__global__ void oproj_kernel(const _Float16* __restrict__ attnh,
                             const _Float16* __restrict__ wo,
                             float* __restrict__ out) {
  const int lane = threadIdx.x;
  const int m0 = blockIdx.y * 16;
  const int n0 = blockIdx.x * 64;
  v8f acc[4] = {};
  for (int k0 = 0; k0 < D_MODEL; k0 += 32) {
    const v16h a = load_a_frag(attnh + (size_t)m0 * D_MODEL + k0, D_MODEL, lane);
#pragma unroll
    for (int j = 0; j < 4; ++j) {
      const v16h b = load_bt_frag(wo + (size_t)(n0 + j * 16) * D_MODEL + k0, D_MODEL, lane);
      acc[j] = WMMA_F16(a, b, acc[j]);
    }
  }
  const int hg = lane >> 4, cb = lane & 15;
#pragma unroll
  for (int r = 0; r < 8; ++r) {
#pragma unroll
    for (int j = 0; j < 4; ++j) {
      out[(size_t)(m0 + r + 8 * hg) * D_MODEL + n0 + j * 16 + cb] = acc[j][r];
    }
  }
}

// ---------------------------------------------------------------------------
extern "C" void kernel_launch(void* const* d_in, const int* in_sizes, int n_in,
                              void* d_out, int out_size, void* d_ws, size_t ws_size,
                              hipStream_t stream) {
  (void)in_sizes; (void)n_in; (void)out_size; (void)ws_size;
  const float* x  = (const float*)d_in[0];
  const int*   tp = (const int*)d_in[1];
  const float* Wq = (const float*)d_in[2];
  const float* Wk = (const float*)d_in[3];
  const float* Wv = (const float*)d_in[4];
  const float* Wo = (const float*)d_in[5];
  float* out = (float*)d_out;

  _Float16* ws = (_Float16*)d_ws;
  const size_t NX = (size_t)NTOK * D_MODEL;     // 8,388,608
  const size_t NW = (size_t)D_MODEL * D_MODEL;  // 1,048,576
  _Float16* xh    = ws;
  _Float16* wqh   = xh + NX;
  _Float16* wkh   = wqh + NW;
  _Float16* wvh   = wkh + NW;
  _Float16* woh   = wvh + NW;
  _Float16* Qh    = woh + NW;
  _Float16* Kh    = Qh + NX;
  _Float16* Vt    = Kh + NX;
  _Float16* attnh = Vt + NX;   // total ~88 MB of workspace

  cvt_f32_f16<<<2048, 256, 0, stream>>>(x,  xh,  (int)NX);
  cvt_f32_f16<<<512,  256, 0, stream>>>(Wq, wqh, (int)NW);
  cvt_f32_f16<<<512,  256, 0, stream>>>(Wk, wkh, (int)NW);
  cvt_f32_f16<<<512,  256, 0, stream>>>(Wv, wvh, (int)NW);
  cvt_f32_f16<<<512,  256, 0, stream>>>(Wo, woh, (int)NW);

  qkv_rope_kernel<<<dim3(D_MODEL / 64, NTOK / 16, 3), 32, 0, stream>>>(
      xh, wqh, wkh, wvh, tp, Qh, Kh, Vt);

  attn_kernel<<<dim3(SEQ / 16, BATCH * NUM_HEADS), 32, 0, stream>>>(
      Qh, Kh, Vt, attnh);

  oproj_kernel<<<dim3(D_MODEL / 64, NTOK / 16), 32, 0, stream>>>(attnh, woh, out);
}
